// Deep_MT_GNN_DGF_17987323035754
// MI455X (gfx1250) — matrix operand order repaired and checked by reading, not compiled, not run on hardware
//
#include <hip/hip_runtime.h>
#include <hip/hip_bf16.h>

// ============================================================================
// Deep_MT_GNN_DGF forward for MI455X (gfx1250, wave32, WMMA f16->f32).
//
// Assumed input flattening (jax tree order: top dict insertion order for the
// 4 top entries, then params flattened with dicts sorted by key, lists in
// order; _lin -> {b,w}, _ln -> {b,g}, gnn -> {adj,ln{b,g},proj{b,w}},
// dgf -> {ln{b,g},log_sigmas,proj{b,w}}):
//   0 maps [B,5,32,32]   1 stats [B,32,7]   2 peri [B,55]
//   3 a1.b  4 a1.w  5 a2.b  6 a2.w  7 cnn.b  8 cnn.w
//   9 conv1.b 10 conv1.w 11 conv2.b 12 conv2.w
//  13 dgf0.ln.b 14 dgf0.ln.g 15 dgf0.log_sigmas 16 dgf0.proj.b 17 dgf0.proj.w
//  18 dgf1.ln.b 19 dgf1.ln.g 20 dgf1.log_sigmas 21 dgf1.proj.b 22 dgf1.proj.w
//  23 eeg.b 24 eeg.w
//  25 gnn0.adj 26 gnn0.ln.b 27 gnn0.ln.g 28 gnn0.proj.b 29 gnn0.proj.w
//  30 gnn1.adj 31 gnn1.ln.b 32 gnn1.ln.g 33 gnn1.proj.b 34 gnn1.proj.w
//  35 modality_emb 36 peri1.b 37 peri1.w 38 peri2.b 39 peri2.w
//  40 peri_ln.b 41 peri_ln.g
//  42..51 region{0..4}.{b,w}   52 v1.b 53 v1.w 54 v2.b 55 v2.w
//
// Output: v [B,2] then a [B,2], f32, concatenated flat.
// ============================================================================

typedef __attribute__((ext_vector_type(16))) _Float16 v16h;
typedef __attribute__((ext_vector_type(8)))  _Float16 v8h;
typedef __attribute__((ext_vector_type(8)))  float    v8f;

__device__ __forceinline__ float elu_f(float x) { return x > 0.f ? x : __expf(x) - 1.f; }

// ---------------------------------------------------------------------------
// CDNA5 async global->LDS staging (GLOBAL_LOAD_ASYNC_TO_LDS_B128, ASYNCcnt),
// with a plain-copy fallback if the builtin is unavailable on this toolchain.
// Probed signature: (v4i __device__* src, v4i __shared__* dst, Ii off, Ii cpol)
// ---------------------------------------------------------------------------
#if defined(__gfx1250__) && __has_builtin(__builtin_amdgcn_global_load_async_to_lds_b128)
#define ASYNC_LDS_COPY 1
#else
#define ASYNC_LDS_COPY 0
#endif

typedef int async_v4i __attribute__((vector_size(16)));

__device__ __forceinline__ void async_wait_all() {
#if ASYNC_LDS_COPY
#if __has_builtin(__builtin_amdgcn_s_wait_asynccnt)
  __builtin_amdgcn_s_wait_asynccnt(0);
#else
  asm volatile("s_wait_asynccnt 0x0" ::: "memory");
#endif
#endif
}

// Copy `bytes` (multiple of 16, both pointers 16B aligned) global -> LDS,
// cooperatively across `nthr` threads.
__device__ __forceinline__ void stage_to_lds(void* dstLds, const void* srcGlobal,
                                             int bytes, int tid, int nthr) {
#if ASYNC_LDS_COPY
  __attribute__((address_space(1))) async_v4i* s =
      (__attribute__((address_space(1))) async_v4i*)srcGlobal;
  __attribute__((address_space(3))) async_v4i* d =
      (__attribute__((address_space(3))) async_v4i*)dstLds;
  for (int i = tid; i < bytes / 16; i += nthr)
    __builtin_amdgcn_global_load_async_to_lds_b128(s + i, d + i, 0, 0);
#else
  const uint4* s = (const uint4*)srcGlobal;
  uint4* d = (uint4*)dstLds;
  for (int i = tid; i < bytes / 16; i += nthr) d[i] = s[i];
#endif
}

// A fragment (16x32 f16, M rows in memory row-major, leading-dim ld halves).
__device__ __forceinline__ v16h frag_a_load(const _Float16* base, int ld, int lane) {
  int m  = lane & 15;
  int kh = (lane >> 4) << 3;                 // 0 or 8
  const _Float16* p = base + m * ld + kh;    // chunks [kh..kh+7], [kh+16..kh+23]
  v8h c0 = *(const v8h*)(p);
  v8h c1 = *(const v8h*)(p + 16);
  return __builtin_shufflevector(c0, c1, 0,1,2,3,4,5,6,7,8,9,10,11,12,13,14,15);
}

// B fragment (32x16 f16) from N-major storage Bt[n][k], leading-dim ldk halves.
__device__ __forceinline__ v16h frag_b_load(const _Float16* bt, int ldk, int lane) {
  int n  = lane & 15;
  int kb = (lane >> 4) << 3;                 // 0 or 8
  const _Float16* p = bt + n * ldk + kb;     // chunks [kb..kb+7], [kb+16..kb+23]
  v8h c0 = *(const v8h*)(p);
  v8h c1 = *(const v8h*)(p + 16);
  return __builtin_shufflevector(c0, c1, 0,1,2,3,4,5,6,7,8,9,10,11,12,13,14,15);
}

__device__ __forceinline__ v8f wmma_f16(v16h a, v16h b, v8f c) {
  return __builtin_amdgcn_wmma_f32_16x16x32_f16(false, a, false, b, (short)0, c, false, false);
}

// C/D fragment: lane group 0-15 holds M=0..7 (vgpr i -> M=i), N=lane;
//               lane group 16-31 holds M=8..15, N=lane-16.
__device__ __forceinline__ void frag_store_f16(_Float16* base, int ld, int lane, v8f c) {
  int n = lane & 15, mb = (lane >> 4) << 3;
#pragma unroll
  for (int i = 0; i < 8; ++i) base[(mb + i) * ld + n] = (_Float16)c[i];
}
__device__ __forceinline__ void frag_store_f32(float* base, int ld, int lane, v8f c) {
  int n = lane & 15, mb = (lane >> 4) << 3;
#pragma unroll
  for (int i = 0; i < 8; ++i) base[(mb + i) * ld + n] = c[i];
}

// --------------------------------------------------------------------------
// Region index tables
// --------------------------------------------------------------------------
__constant__ int RIDX[32] = {0,1,2,3,4,11,12,13,14,15,  6,7,8,20,21,22,
                             9,10,24,25,  28,29,30,  5,16,17,18,19,23,26,27,31};
__constant__ int RSTART[6] = {0,10,16,20,23,32};

// ============================================================================
// Kernel 0: weight prep — softmax(adj) and f16 transposed weight copies.
// ============================================================================
struct PrepArgs {
  const float* src[13];
  int din[13];                  // K
  int dout[13];                 // N
  unsigned long long dstOff[13];// element offset into h16 pool
  unsigned long long total;
  const float* adjSrc[2];
  float* adjDst;                // ws, [2][32][32] f32
  _Float16* h16;
};

__launch_bounds__(256)
__global__ void prep_kernel(PrepArgs A) {
  long gt = (long)blockIdx.x * blockDim.x + threadIdx.x;
  long stride = (long)gridDim.x * blockDim.x;
  if (gt < 64) {                              // row-softmax of both adj matrices
    int layer = (int)(gt >> 5), row = (int)(gt & 31);
    const float* s = A.adjSrc[layer] + row * 32;
    float mx = s[0];
    for (int j = 1; j < 32; ++j) mx = fmaxf(mx, s[j]);
    float e[32], den = 0.f;
    for (int j = 0; j < 32; ++j) { e[j] = __expf(s[j] - mx); den += e[j]; }
    float inv = 1.f / den;
    float* d = A.adjDst + layer * 1024 + row * 32;
    for (int j = 0; j < 32; ++j) d[j] = e[j] * inv;
  }
  for (long idx = gt; idx < (long)A.total; idx += stride) {
    long r = idx; int mi = 0;
    while (mi < 13) {
      long sz = (long)A.din[mi] * A.dout[mi];
      if (r < sz) break;
      r -= sz; ++mi;
    }
    int K = A.din[mi], N = A.dout[mi];
    long k = r / N, n = r - k * N;
    A.h16[A.dstOff[mi] + n * (long)K + k] = (_Float16)A.src[mi][k * (long)N + n];
  }
}

// ============================================================================
// Kernel 1: EEG projection + 2 GNN layers + region pooling. 4 samples/block,
// 128 threads (4 waves). All 64-dim contractions via WMMA.
// ============================================================================
struct GnnArgs {
  const float* stats;     // [B,32,7]
  const float* eegW;      // [7,64]
  const float* eegB;      // [64]
  const float* adjS;      // ws [2][32][32] f32 (softmaxed)
  const _Float16* gnnWt;  // ws [2][64][64]  (W^T: [n][k])
  const float* projB[2];
  const float* lnG[2];
  const float* lnB[2];
  const _Float16* regWt;  // ws [5][64][64]
  const float* regB[5];
  const float* emb;       // [21,64]
  float* x;               // [B,21,64]
};

__launch_bounds__(128)
__global__ void gnn_kernel(GnnArgs A) {
  __shared__ alignas(16) _Float16 hs[128 * 64];    // h rows (sample s*32+node)
  __shared__ alignas(16) _Float16 hT[4 * 64 * 32]; // per-sample transpose [s][d][n]
  __shared__ alignas(16) _Float16 ts[128 * 64];    // t rows / y rows / region-A
  __shared__ alignas(16) _Float16 Wt[64 * 64];
  __shared__ alignas(16) _Float16 adjf[32 * 32];
  int tid = threadIdx.x, lane = tid & 31, wv = tid >> 5;
  long s0 = (long)blockIdx.x * 4;

  // ---- eeg_proj (K=7, VALU) ----
  {
    int r = tid, s = r >> 5, n = r & 31;
    const float* st = A.stats + ((s0 + s) * 32 + n) * 7;
    float sv[7];
#pragma unroll
    for (int k = 0; k < 7; ++k) sv[k] = st[k];
    for (int d = 0; d < 64; ++d) {
      float acc = A.eegB[d];
#pragma unroll
      for (int k = 0; k < 7; ++k) acc += sv[k] * A.eegW[k * 64 + d];
      _Float16 h = (_Float16)acc;
      hs[r * 64 + d] = h;
      hT[s * 2048 + d * 32 + n] = h;
    }
  }
  __syncthreads();

  // ---- 2 GNN layers ----
  for (int p = 0; p < 2; ++p) {
    stage_to_lds(Wt, A.gnnWt + p * 4096, 4096 * 2, tid, 128);     // async f16 W^T
    for (int i = tid; i < 1024; i += 128) adjf[i] = (_Float16)A.adjS[p * 1024 + i];
    async_wait_all();
    __syncthreads();

    // t = softmax(adj) @ h   (wave wv owns sample wv)
    {
      int s = wv;
      for (int mt = 0; mt < 2; ++mt)
        for (int nt = 0; nt < 4; ++nt) {
          v16h a = frag_a_load(adjf + mt * 16 * 32, 32, lane);
          v16h b = frag_b_load(hT + s * 2048 + nt * 16 * 32, 32, lane);
          v8f c = {};
          c = wmma_f16(a, b, c);
          frag_store_f16(ts + (s * 32 + mt * 16) * 64 + nt * 16, 64, lane, c);
        }
    }
    __syncthreads();

    // u = t @ W + b; y = elu(u) + h  -> ts (f16)
    for (int half = 0; half < 2; ++half) {
      int mt = wv * 2 + half;
      v8f acc[4];
      v8f zero = {};
#pragma unroll
      for (int nt = 0; nt < 4; ++nt) acc[nt] = zero;
      for (int ks = 0; ks < 2; ++ks) {
        v16h a = frag_a_load(ts + mt * 16 * 64 + ks * 32, 64, lane);
#pragma unroll
        for (int nt = 0; nt < 4; ++nt) {
          v16h b = frag_b_load(Wt + (nt * 16) * 64 + ks * 32, 64, lane);
          acc[nt] = wmma_f16(a, b, acc[nt]);
        }
      }
      int mb = (lane >> 4) << 3, nn = lane & 15;
      for (int nt = 0; nt < 4; ++nt) {
        int col = nt * 16 + nn;
        float bias = A.projB[p][col];
#pragma unroll
        for (int i = 0; i < 8; ++i) {
          int row = mt * 16 + mb + i;
          float y = elu_f(acc[nt][i] + bias) + (float)hs[row * 64 + col];
          ts[row * 64 + col] = (_Float16)y;
        }
      }
    }
    __syncthreads();

    // LayerNorm per row -> hs, hT
    {
      int r = tid;
      float m = 0.f;
      for (int d = 0; d < 64; ++d) m += (float)ts[r * 64 + d];
      m *= (1.f / 64.f);
      float v = 0.f;
      for (int d = 0; d < 64; ++d) { float q = (float)ts[r * 64 + d] - m; v += q * q; }
      v *= (1.f / 64.f);
      float rv = 1.f / sqrtf(v + 1e-5f);
      int s = r >> 5, n = r & 31;
      for (int d = 0; d < 64; ++d) {
        float o = ((float)ts[r * 64 + d] - m) * rv * A.lnG[p][d] + A.lnB[p][d];
        _Float16 h = (_Float16)o;
        hs[r * 64 + d] = h;
        hT[s * 2048 + d * 32 + n] = h;
      }
    }
    __syncthreads();
  }

  // ---- region pooling: A tiles [region][16(samp,pad)][64] in ts ----
  for (int i = tid; i < 5 * 16 * 64; i += 128) ts[i] = (_Float16)0.f;
  __syncthreads();
  for (int i = tid; i < 5 * 4 * 64; i += 128) {
    int r = i >> 8, rem = i & 255, s = rem >> 6, d = rem & 63;
    int st = RSTART[r], en = RSTART[r + 1];
    float acc = 0.f;
    for (int j = st; j < en; ++j) acc += (float)hs[(s * 32 + RIDX[j]) * 64 + d];
    ts[(r * 16 + s) * 64 + d] = (_Float16)(acc / (float)(en - st));
  }
  __syncthreads();

  for (int r = 0; r < 5; ++r) {
    stage_to_lds(Wt, A.regWt + r * 4096, 4096 * 2, tid, 128);
    async_wait_all();
    __syncthreads();
    {
      int nt = wv;                       // 4 waves = 4 column tiles
      v8f c = {};
      for (int ks = 0; ks < 2; ++ks) {
        v16h a = frag_a_load(ts + r * 16 * 64 + ks * 32, 64, lane);
        v16h b = frag_b_load(Wt + nt * 16 * 64 + ks * 32, 64, lane);
        c = wmma_f16(a, b, c);
      }
      int mb = (lane >> 4) << 3, nn = lane & 15;
      if (mb == 0) {                     // only rows 0..3 hold real samples
        int col = nt * 16 + nn;
        float bias = A.regB[r][col];
        float em = A.emb[r * 64 + col];
#pragma unroll
        for (int i = 0; i < 4; ++i) {
          float v = elu_f(c[i] + bias) + em;
          A.x[((s0 + i) * 21 + r) * 64 + col] = v;
        }
      }
    }
    __syncthreads();
  }
}

// ============================================================================
// Kernel 2: fused CNN branch (conv1+IN+ELU+maxpool, conv2+IN+ELU+avgpool).
// One block per sample; everything stays in LDS. (max commutes with the
// monotonic IN+ELU, so pool raw conv1 values after computing pre-pool stats.)
// ============================================================================
struct CnnArgs {
  const float* maps;                // [B,5,32,32]
  const float* w1; const float* b1; // [16,5,3,3],[16]
  const float* w2; const float* b2; // [32,16,3,3],[32]
  float* cpool;                     // [B,128]
};

__launch_bounds__(256)
__global__ void cnn_kernel(CnnArgs A) {
  __shared__ alignas(16) float in5[5 * 1024];
  __shared__ alignas(16) float w1s[720];
  __shared__ alignas(16) float c1s[16 * 256];
  __shared__ alignas(16) float w2s[4608];
  __shared__ float stat[64];
  __shared__ float pacc[128];
  int t = threadIdx.x;
  long b = blockIdx.x;
  stage_to_lds(in5, A.maps + b * 5120, 5120 * 4, t, 256);   // async 20KB tile
  stage_to_lds(w1s, A.w1, 720 * 4, t, 256);
  stage_to_lds(w2s, A.w2, 4608 * 4, t, 256);
  if (t < 64) stat[t] = 0.f;
  if (t < 128) pacc[t] = 0.f;
  async_wait_all();
  __syncthreads();

  // conv1: thread t owns pooling window (wy,wx) of every channel
  int wy = t >> 4, wx = t & 15;
  float mx[16];
  for (int c = 0; c < 16; ++c) {
    float bias = A.b1[c];
    float vmax = -1e30f, s = 0.f, sq = 0.f;
#pragma unroll
    for (int dy = 0; dy < 2; ++dy)
#pragma unroll
      for (int dx = 0; dx < 2; ++dx) {
        int y = wy * 2 + dy, x = wx * 2 + dx;
        float acc = bias;
        for (int ic = 0; ic < 5; ++ic)
#pragma unroll
          for (int ky = 0; ky < 3; ++ky) {
            int yy = y + ky - 1;
            if (yy < 0 || yy > 31) continue;
#pragma unroll
            for (int kx = 0; kx < 3; ++kx) {
              int xx = x + kx - 1;
              if (xx < 0 || xx > 31) continue;
              acc += in5[ic * 1024 + yy * 32 + xx] * w1s[((c * 5 + ic) * 3 + ky) * 3 + kx];
            }
          }
        vmax = fmaxf(vmax, acc); s += acc; sq += acc * acc;
      }
    mx[c] = vmax;
    atomicAdd(&stat[c], s);
    atomicAdd(&stat[32 + c], sq);
  }
  __syncthreads();
  for (int c = 0; c < 16; ++c) {
    float m = stat[c] * (1.f / 1024.f);
    float v = stat[32 + c] * (1.f / 1024.f) - m * m;
    float r = 1.f / sqrtf(v + 1e-5f);
    c1s[c * 256 + t] = elu_f((mx[c] - m) * r);
  }
  __syncthreads();
  if (t < 64) stat[t] = 0.f;
  __syncthreads();

  // conv2: thread t owns pixel t of every channel
  int py = t >> 4, px = t & 15;
  float v2r[32];
  for (int c = 0; c < 32; ++c) {
    float acc = A.b2[c];
    for (int ic = 0; ic < 16; ++ic)
#pragma unroll
      for (int ky = 0; ky < 3; ++ky) {
        int yy = py + ky - 1;
        if (yy < 0 || yy > 15) continue;
#pragma unroll
        for (int kx = 0; kx < 3; ++kx) {
          int xx = px + kx - 1;
          if (xx < 0 || xx > 15) continue;
          acc += c1s[ic * 256 + yy * 16 + xx] * w2s[((c * 16 + ic) * 3 + ky) * 3 + kx];
        }
      }
    v2r[c] = acc;
    atomicAdd(&stat[c], acc);
    atomicAdd(&stat[32 + c], acc * acc);
  }
  __syncthreads();
  int q = ((py >> 3) << 1) | (px >> 3);
  for (int c = 0; c < 32; ++c) {
    float m = stat[c] * (1.f / 256.f);
    float v = stat[32 + c] * (1.f / 256.f) - m * m;
    float r = 1.f / sqrtf(v + 1e-5f);
    atomicAdd(&pacc[c * 4 + q], elu_f((v2r[c] - m) * r));
  }
  __syncthreads();
  if (t < 128) A.cpool[b * 128 + t] = pacc[t] * (1.f / 64.f);
}

// ============================================================================
// Kernel 3: peri1 (K=55 VALU) + LN + ELU
// ============================================================================
struct PeriArgs {
  const float* peri; const float* w1; const float* b1;
  const float* g; const float* bt; float* out;
};
__launch_bounds__(128)
__global__ void peri1_kernel(PeriArgs A) {
  __shared__ float p[55];
  __shared__ float st[2];
  int t = threadIdx.x;
  long b = blockIdx.x;
  if (t < 55) p[t] = A.peri[b * 55 + t];
  if (t < 2) st[t] = 0.f;
  __syncthreads();
  float acc = A.b1[t];
  for (int k = 0; k < 55; ++k) acc += p[k] * A.w1[k * 128 + t];
  atomicAdd(&st[0], acc);
  atomicAdd(&st[1], acc * acc);
  __syncthreads();
  float m = st[0] * (1.f / 128.f);
  float v = st[1] * (1.f / 128.f) - m * m;
  float r = 1.f / sqrtf(v + 1e-5f);
  A.out[b * 128 + t] = elu_f((acc - m) * r * A.g[t] + A.bt[t]);
}

// ============================================================================
// Kernel 4: [B,128] @ W[128,512] + b, reshape to 8 nodes, += emb -> x.
// WMMA GEMM, 128 rows x 512 cols per block (8 waves, 1 M-tile each).
// ============================================================================
struct NodegenArgs {
  const float* src;      // [B,128]
  const _Float16* Wt;    // [512][128]
  const float* bias;     // [512]
  const float* emb;      // [21,64]
  int nodeBase;
  float* x;              // [B,21,64]
};
__launch_bounds__(256)
__global__ void nodegen_kernel(NodegenArgs A) {
  __shared__ alignas(16) _Float16 a16[128 * 128];
  int t = threadIdx.x, lane = t & 31, wv = t >> 5;
  long r0 = (long)blockIdx.x * 128;
  for (int i = t; i < 128 * 128; i += 256) a16[i] = (_Float16)A.src[r0 * 128 + i];
  __syncthreads();
  int mt = wv;
  int mb = (lane >> 4) << 3, nn = lane & 15;
  for (int nt = 0; nt < 32; ++nt) {
    if (nt + 1 < 32)
      __builtin_prefetch(A.Wt + ((nt + 1) * 16 + (lane & 15)) * 128, 0, 1);
    v8f c = {};
    for (int ks = 0; ks < 4; ++ks) {
      v16h a = frag_a_load(a16 + mt * 16 * 128 + ks * 32, 128, lane);
      v16h b = frag_b_load(A.Wt + (nt * 16) * 128 + ks * 32, 128, lane);
      c = wmma_f16(a, b, c);
    }
    int col = nt * 16 + nn;
    int node = A.nodeBase + (col >> 6), d = col & 63;
    float add = A.bias[col] + A.emb[node * 64 + d];
#pragma unroll
    for (int i = 0; i < 8; ++i) {
      long row = r0 + mt * 16 + mb + i;
      A.x[(row * 21 + node) * 64 + d] = c[i] + add;
    }
  }
}

// ============================================================================
// Kernel 5: DGF layer. 2 samples/block (1 per wave). Gram matrix,
// adj@x and proj all on WMMA; adjacency/exp/LN on VALU.
// ============================================================================
struct DgfArgs {
  float* x;               // [B,21,64] (in/out)
  const _Float16* Wt;     // [64][64]
  const float* projB; const float* lnG; const float* lnB;
  const float* logSig;    // [8]
  _Float16* xf16;         // [B,1344] (written when writeF16)
  int writeF16;
};
__launch_bounds__(64)
__global__ void dgf_kernel(DgfArgs A) {
  __shared__ alignas(16) _Float16 xs[2][32 * 64];
  __shared__ alignas(16) _Float16 xT[2][64 * 32];
  __shared__ alignas(16) _Float16 tf[2][32 * 64];
  __shared__ alignas(16) _Float16 adjf[2][32 * 32];
  __shared__ alignas(16) float    gram[2][32 * 32];
  __shared__ float    sq[2][32];
  __shared__ alignas(16) _Float16 Wt[64 * 64];
  int t = threadIdx.x, lane = t & 31, wv = t >> 5;
  long s = (long)blockIdx.x * 2 + wv;
  float* xp = A.x + s * 21 * 64;

  stage_to_lds(Wt, A.Wt, 4096 * 2, t, 64);           // async f16 W^T
  for (int i = lane; i < 32 * 64; i += 32) xs[wv][i] = (_Float16)0.f;
  for (int i = lane; i < 64 * 32; i += 32) xT[wv][i] = (_Float16)0.f;
  for (int i = lane; i < 32 * 32; i += 32) adjf[wv][i] = (_Float16)0.f;
  __syncthreads();
  for (int i = lane; i < 21 * 64; i += 32) {
    int n = i >> 6, d = i & 63;
    float v = xp[i];
    xs[wv][n * 64 + d] = (_Float16)v;
    xT[wv][d * 32 + n] = (_Float16)v;
  }
  async_wait_all();
  __syncthreads();
  {
    float a0 = 0.f;
    if (lane < 21)
      for (int d = 0; d < 64; ++d) { float q = (float)xs[wv][lane * 64 + d]; a0 += q * q; }
    sq[wv][lane] = (lane < 21) ? a0 : 0.f;
  }
  __syncthreads();

  // gram = x @ x^T  (B operand is xs itself: N-major node x dim)
  for (int mt = 0; mt < 2; ++mt)
    for (int nt = 0; nt < 2; ++nt) {
      v8f c = {};
      for (int ks = 0; ks < 2; ++ks) {
        v16h a = frag_a_load(xs[wv] + mt * 16 * 64 + ks * 32, 64, lane);
        v16h b = frag_b_load(xs[wv] + nt * 16 * 64 + ks * 32, 64, lane);
        c = wmma_f16(a, b, c);
      }
      frag_store_f32(gram[wv] + mt * 16 * 32 + nt * 16, 32, lane, c);
    }
  __syncthreads();

  // RBF adjacency (mean over 8 sigmas)
  {
    float dinv[8];
#pragma unroll
    for (int j = 0; j < 8; ++j) dinv[j] = 1.f / (2.f * __expf(2.f * A.logSig[j]) + 1e-6f);
    for (int e = lane; e < 441; e += 32) {
      int n = e / 21, m = e - n * 21;
      float d2 = fmaxf(sq[wv][n] + sq[wv][m] - 2.f * gram[wv][n * 32 + m], 0.f);
      float acc = 0.f;
#pragma unroll
      for (int j = 0; j < 8; ++j) acc += __expf(-d2 * dinv[j]);
      adjf[wv][n * 32 + m] = (_Float16)(acc * 0.125f);
    }
  }
  __syncthreads();

  // t = adj @ x
  for (int mt = 0; mt < 2; ++mt)
    for (int nt = 0; nt < 4; ++nt) {
      v16h a = frag_a_load(adjf[wv] + mt * 16 * 32, 32, lane);
      v16h b = frag_b_load(xT[wv] + nt * 16 * 32, 32, lane);
      v8f c = {};
      c = wmma_f16(a, b, c);
      frag_store_f16(tf[wv] + mt * 16 * 64 + nt * 16, 64, lane, c);
    }
  __syncthreads();

  // u = t @ W + b; y = elu(u) + x  -> tf
  {
    int mb = (lane >> 4) << 3, nn = lane & 15;
    for (int mt = 0; mt < 2; ++mt) {
      v8f acc[4];
      v8f zero = {};
#pragma unroll
      for (int nt = 0; nt < 4; ++nt) acc[nt] = zero;
      for (int ks = 0; ks < 2; ++ks) {
        v16h a = frag_a_load(tf[wv] + mt * 16 * 64 + ks * 32, 64, lane);
#pragma unroll
        for (int nt = 0; nt < 4; ++nt) {
          v16h b = frag_b_load(Wt + nt * 16 * 64 + ks * 32, 64, lane);
          acc[nt] = wmma_f16(a, b, acc[nt]);
        }
      }
      for (int nt = 0; nt < 4; ++nt) {
        int col = nt * 16 + nn;
        float bias = A.projB[col];
#pragma unroll
        for (int i = 0; i < 8; ++i) {
          int row = mt * 16 + mb + i;
          float y = elu_f(acc[nt][i] + bias) + (float)xs[wv][row * 64 + col];
          tf[wv][row * 64 + col] = (_Float16)y;
        }
      }
    }
  }
  __syncthreads();

  // LayerNorm + writeback
  if (lane < 21) {
    int r = lane;
    float m = 0.f;
    for (int d = 0; d < 64; ++d) m += (float)tf[wv][r * 64 + d];
    m *= (1.f / 64.f);
    float v = 0.f;
    for (int d = 0; d < 64; ++d) { float q = (float)tf[wv][r * 64 + d] - m; v += q * q; }
    v *= (1.f / 64.f);
    float rv = 1.f / sqrtf(v + 1e-5f);
    for (int d = 0; d < 64; ++d) {
      float o = ((float)tf[wv][r * 64 + d] - m) * rv * A.lnG[d] + A.lnB[d];
      xp[r * 64 + d] = o;
      if (A.writeF16) A.xf16[s * 1344 + r * 64 + d] = (_Float16)o;
    }
  }
}

// ============================================================================
// Kernel 6: heads. WMMA GEMM [B,1344]@[1344,64] + ELU, then 64->2 on VALU.
// blockIdx.y selects v/a head. 128 rows/block, wave per 16-row tile.
// ============================================================================
struct HeadArgs {
  const _Float16* xf16;    // [B,1344]
  const _Float16* Wt[2];   // [64][1344]
  const float* b1[2];      // [64]
  const float* w2[2];      // [64,2]
  const float* b2[2];      // [2]
  float* out;              // v then a
  long B;
};
__launch_bounds__(256)
__global__ void head_kernel(HeadArgs A) {
  __shared__ alignas(16) float hid[8][16 * 64];
  int t = threadIdx.x, lane = t & 31, wv = t >> 5;
  int h = blockIdx.y;
  long r0 = (long)blockIdx.x * 128 + wv * 16;
  const _Float16* xa = A.xf16 + r0 * 1344;
  const _Float16* wt = A.Wt[h];
  int mb = (lane >> 4) << 3, nn = lane & 15;
  v8f acc[4];
  v8f zero = {};
#pragma unroll
  for (int nt = 0; nt < 4; ++nt) acc[nt] = zero;
  for (int ks = 0; ks < 42; ++ks) {
    if (ks + 2 < 42) {
      __builtin_prefetch(wt + (lane & 15) * 1344 + (ks + 2) * 32, 0, 1);
      __builtin_prefetch(xa + (lane & 15) * 1344 + (ks + 2) * 32, 0, 1);
    }
    v16h a = frag_a_load(xa + ks * 32, 1344, lane);
#pragma unroll
    for (int nt = 0; nt < 4; ++nt) {
      v16h b = frag_b_load(wt + nt * 16 * 1344 + ks * 32, 1344, lane);
      acc[nt] = wmma_f16(a, b, acc[nt]);
    }
  }
  for (int nt = 0; nt < 4; ++nt) {
    int col = nt * 16 + nn;
    float bias = A.b1[h][col];
#pragma unroll
    for (int i = 0; i < 8; ++i) hid[wv][(mb + i) * 64 + col] = elu_f(acc[nt][i] + bias);
  }
  __syncthreads();
  {
    int row = lane >> 1, col = lane & 1;
    float a2 = A.b2[h][col];
    for (int k = 0; k < 64; ++k) a2 += hid[wv][row * 64 + k] * A.w2[h][k * 2 + col];
    A.out[(long)h * A.B * 2 + (r0 + row) * 2 + col] = a2;
  }
}

// ============================================================================
// Host launcher
// ============================================================================
extern "C" void kernel_launch(void* const* d_in, const int* in_sizes, int n_in,
                              void* d_out, int out_size, void* d_ws, size_t ws_size,
                              hipStream_t stream) {
  const float* maps  = (const float*)d_in[0];
  const float* stats = (const float*)d_in[1];
  const float* peri  = (const float*)d_in[2];
  auto F = [&](int i) { return (const float*)d_in[i]; };
  long B = (long)in_sizes[1] / (32 * 7);

  char* ws = (char*)d_ws;
  float* adjS = (float*)ws;                              // 2*32*32 f32 = 8192 B
  _Float16* h16 = (_Float16*)(ws + 8192);                // f16 weight pool
  // f16 pool element offsets
  const size_t O_GNNWT = 0;        // 2*4096
  const size_t O_DGFWT = 8192;     // 2*4096
  const size_t O_REGWT = 16384;    // 5*4096
  const size_t O_CNNWT = 36864;    // 512*128
  const size_t O_P2WT  = 102400;   // 512*128
  const size_t O_V1WT  = 167936;   // 64*1344
  const size_t O_A1WT  = 253952;   // 64*1344
  const size_t H16_TOTAL = 339968;
  size_t actOff = 8192 + H16_TOTAL * 2;                  // 688128 B
  float* x       = (float*)(ws + actOff);                // [B,21,64] f32
  _Float16* xf16 = (_Float16*)(ws + actOff + (size_t)B * 21 * 64 * 4);
  float* cpool   = (float*)((char*)xf16 + (size_t)B * 1344 * 2);
  float* pperi   = cpool + (size_t)B * 128;

  // ---- prep: softmax(adj) + f16 transposed weights ----
  {
    PrepArgs pa;
    const int srcIdx[13] = {29, 34, 17, 22, 43, 45, 47, 49, 51, 8, 39, 53, 4};
    const int dins[13]   = {64, 64, 64, 64, 64, 64, 64, 64, 64, 128, 128, 1344, 1344};
    const int douts[13]  = {64, 64, 64, 64, 64, 64, 64, 64, 64, 512, 512, 64, 64};
    unsigned long long off = 0;
    for (int i = 0; i < 13; ++i) {
      pa.src[i] = F(srcIdx[i]); pa.din[i] = dins[i]; pa.dout[i] = douts[i];
      pa.dstOff[i] = off; off += (unsigned long long)dins[i] * douts[i];
    }
    pa.total = off;
    pa.adjSrc[0] = F(25); pa.adjSrc[1] = F(30);
    pa.adjDst = adjS; pa.h16 = h16;
    prep_kernel<<<512, 256, 0, stream>>>(pa);
  }

  // ---- GNN branch -> x nodes 0..4 ----
  {
    GnnArgs ga;
    ga.stats = stats; ga.eegW = F(24); ga.eegB = F(23);
    ga.adjS = adjS; ga.gnnWt = h16 + O_GNNWT;
    ga.projB[0] = F(28); ga.projB[1] = F(33);
    ga.lnG[0] = F(27);  ga.lnG[1] = F(32);
    ga.lnB[0] = F(26);  ga.lnB[1] = F(31);
    ga.regWt = h16 + O_REGWT;
    ga.regB[0] = F(42); ga.regB[1] = F(44); ga.regB[2] = F(46);
    ga.regB[3] = F(48); ga.regB[4] = F(50);
    ga.emb = F(35); ga.x = x;
    gnn_kernel<<<(int)(B / 4), 128, 0, stream>>>(ga);
  }

  // ---- CNN branch -> cpool ----
  {
    CnnArgs ca;
    ca.maps = maps; ca.w1 = F(10); ca.b1 = F(9); ca.w2 = F(12); ca.b2 = F(11);
    ca.cpool = cpool;
    cnn_kernel<<<(int)B, 256, 0, stream>>>(ca);
  }

  // ---- peri branch stage 1 -> pperi ----
  {
    PeriArgs pe;
    pe.peri = peri; pe.w1 = F(37); pe.b1 = F(36); pe.g = F(41); pe.bt = F(40);
    pe.out = pperi;
    peri1_kernel<<<(int)B, 128, 0, stream>>>(pe);
  }

  // ---- node generators (WMMA GEMM 128->512) -> x nodes 5..12 and 13..20 ----
  {
    NodegenArgs na;
    na.src = cpool; na.Wt = h16 + O_CNNWT; na.bias = F(7); na.emb = F(35);
    na.nodeBase = 5; na.x = x;
    nodegen_kernel<<<(int)(B / 128), 256, 0, stream>>>(na);
    na.src = pperi; na.Wt = h16 + O_P2WT; na.bias = F(38);
    na.nodeBase = 13;
    nodegen_kernel<<<(int)(B / 128), 256, 0, stream>>>(na);
  }

  // ---- 2 DGF layers ----
  {
    DgfArgs da;
    da.x = x; da.xf16 = xf16;
    da.Wt = h16 + O_DGFWT; da.projB = F(16); da.lnG = F(14); da.lnB = F(13);
    da.logSig = F(15); da.writeF16 = 0;
    dgf_kernel<<<(int)(B / 2), 64, 0, stream>>>(da);
    da.Wt = h16 + O_DGFWT + 4096; da.projB = F(21); da.lnG = F(19); da.lnB = F(18);
    da.logSig = F(20); da.writeF16 = 1;
    dgf_kernel<<<(int)(B / 2), 64, 0, stream>>>(da);
  }

  // ---- heads -> d_out (v then a) ----
  {
    HeadArgs ha;
    ha.xf16 = xf16;
    ha.Wt[0] = h16 + O_V1WT; ha.Wt[1] = h16 + O_A1WT;
    ha.b1[0] = F(52); ha.b1[1] = F(3);
    ha.w2[0] = F(55); ha.w2[1] = F(6);
    ha.b2[0] = F(54); ha.b2[1] = F(5);
    ha.out = (float*)d_out; ha.B = B;
    head_kernel<<<dim3((unsigned)(B / 128), 2), 256, 0, stream>>>(ha);
  }
}
